// RPN_9191230013520
// MI455X (gfx1250) — matrix operand order repaired and checked by reading, not compiled, not run on hardware
//
#include <hip/hip_runtime.h>
#include <hip/hip_bf16.h>

// ---------------- types ----------------
typedef __attribute__((ext_vector_type(16))) __bf16 v16bf;
typedef __attribute__((ext_vector_type(8)))  float  v8f;

struct U32x2 { uint4 a, b; };   // 32 bytes, bit_cast target for v16bf

__device__ __forceinline__ __bf16 f2bf(float f) {
    unsigned u = __builtin_bit_cast(unsigned, f);
    unsigned r = (u + 0x7FFFu + ((u >> 16) & 1u)) >> 16;
    unsigned short s = (unsigned short)r;
    return __builtin_bit_cast(__bf16, s);
}

// ---- CDNA5 async global->LDS copy (ASYNCcnt-tracked, ISA 15.18.3 op 98) ----
// LDS byte address = low 32 bits of the generic pointer (aperture mapping).
__device__ __forceinline__ void async_cp16(void* lds, const void* g) {
    unsigned l = (unsigned)(unsigned long long)lds;
    asm volatile("global_load_async_to_lds_b128 %0, %1, off"
                 :: "v"(l), "v"(g) : "memory");
}

#define HW 128
#define NPIX 16384          // 128*128
#define CIN 512
#define COUT 512
#define KTOT 4608           // 512*9
#define NANCH 147456        // 16384*9
#define PRE_TOPN 6000
#define POST_TOPN 300
#define SORTN 8192
#define NWORDS 188          // ceil(6000/32)

__device__ const float BASE_ANCH[9][4] = {
    {-84.f,  -40.f,  99.f,  55.f},
    {-176.f, -88.f,  191.f, 103.f},
    {-360.f, -184.f, 375.f, 199.f},
    {-56.f,  -56.f,  71.f,  71.f},
    {-120.f, -120.f, 135.f, 135.f},
    {-248.f, -248.f, 263.f, 263.f},
    {-36.f,  -84.f,  51.f,  99.f},
    {-80.f,  -176.f, 95.f,  191.f},
    {-168.f, -360.f, 183.f, 375.f}
};

// ---------------- init: zero histogram + 32B zero pad source ----------------
__global__ void k_init(unsigned* hist, __bf16* zb) {
    int i = blockIdx.x * 256 + threadIdx.x;
    if (i < 1024) hist[i] = 0u;
    if (i < 16)   zb[i] = __builtin_bit_cast(__bf16, (unsigned short)0);
}

// ---------------- prep kernels ----------------
// feature NCHW f32 -> [pixel][c] bf16 (K-contiguous for GEMM A tiles)
__global__ void k_prep_feat(const float* __restrict__ f, __bf16* __restrict__ featT) {
    size_t i = (size_t)blockIdx.x * 256 + threadIdx.x;
    if (i >= (size_t)NPIX * CIN) return;
    int c = (int)(i & 511);
    int pix = (int)(i >> 9);
    featT[i] = f2bf(f[(size_t)c * NPIX + pix]);
}

// w_rpn_conv OIHW f32 -> pre-swizzled WMMA B fragments:
// wBf[ktile(144)][nt(32)][lane(32)][16]  (B layout: lanes 0-15 K=0..15, 16-31 K=16..31)
__global__ void k_prep_wconv(const float* __restrict__ w, __bf16* __restrict__ wBf) {
    size_t i = (size_t)blockIdx.x * 256 + threadIdx.x;
    if (i >= (size_t)KTOT * COUT) return;
    int n = (int)(i & 511);
    int k = (int)(i >> 9);              // k = (ky*3+kx)*512 + ci
    int plane = k >> 9;
    int ci = k & 511;
    int ky = plane / 3, kx = plane % 3;
    float v = w[(((size_t)n * CIN + ci) * 3 + ky) * 3 + kx];
    int ktile = k >> 5, kin = k & 31;
    int nt = n >> 4, nl = n & 15;
    int lane = (kin >> 4) * 16 + nl;
    size_t dst = ((size_t)(ktile * 32 + nt) * 32 + lane) * 16 + (kin & 15);
    wBf[dst] = f2bf(v);
}

// score(18)+bbox(36)+pad(10) weights -> pre-swizzled fragments
// w2f[kt(16)][nt(4)][lane(32)][16] ; bias2[64] f32
__global__ void k_prep_w2(const float* __restrict__ ws, const float* __restrict__ bs,
                          const float* __restrict__ wb, const float* __restrict__ bb,
                          __bf16* __restrict__ w2f, float* __restrict__ bias2) {
    int i = blockIdx.x * 256 + threadIdx.x;
    if (i >= CIN * 64) return;
    int ci = i >> 6, n = i & 63;
    float v = 0.f;
    if (n < 18)      v = ws[(size_t)n * CIN + ci];
    else if (n < 54) v = wb[(size_t)(n - 18) * CIN + ci];
    int kt = ci >> 5, kin = ci & 31;
    int nt = n >> 4, nl = n & 15;
    int lane = (kin >> 4) * 16 + nl;
    size_t dst = ((size_t)(kt * 4 + nt) * 32 + lane) * 16 + (kin & 15);
    w2f[dst] = f2bf(v);
    if (i < 64) {
        float b = 0.f;
        if (i < 18)      b = bs[i];
        else if (i < 54) b = bb[i - 18];
        bias2[i] = b;
    }
}

// ---------------- big WMMA GEMM : implicit 3x3 conv ----------------
// BM=64 pixels x BN=128 couts, BK=32, 8 waves (4x2), async double-buffered LDS
__global__ __launch_bounds__(256) void k_conv_gemm(
    const __bf16* __restrict__ featT,   // [NPIX][512]
    const __bf16* __restrict__ wBf,     // swizzled fragments
    const float*  __restrict__ bias,    // [512]
    const __bf16* __restrict__ zerobuf, // 32B of zeros
    __bf16* __restrict__ outT)          // [NPIX][512]
{
    __shared__ __attribute__((aligned(32))) __bf16 As[2][64 * 32];       // 2 x 4 KB
    __shared__ __attribute__((aligned(32))) __bf16 Bs[2][8 * 32 * 16];   // 2 x 8 KB
    const int tid  = threadIdx.x;
    const int lane = tid & 31;
    const int wave = tid >> 5;
    const int wm = wave >> 1;           // 0..3
    const int wn = wave & 1;            // 0..1
    const int half = lane >> 4;
    const int l16  = lane & 15;

    const int tileM = blockIdx.x * 64;
    const int tileN = blockIdx.y * 128;

    v8f acc[4] = {};

    const int arow = tid >> 2;          // 0..63
    const int akc  = (tid & 3) * 8;     // 0,8,16,24
    const int apix = tileM + arow;
    const int ay = apix >> 7, ax = apix & 127;

    // per-wave async issue count per stage is exactly 3 (A:1, B:2) -> exact waits
    auto issue = [&](int step, int s) {
        int kk = step * 32;
        int k0 = kk + akc;
        int plane = k0 >> 9;            // 0..8
        int ci = k0 & 511;
        int ky = plane / 3 - 1, kx = plane % 3 - 1;
        int sy = ay + ky, sx = ax + kx;
        const void* asrc = (sy >= 0 && sy < HW && sx >= 0 && sx < HW)
            ? (const void*)(featT + (((size_t)sy * HW + sx) * CIN + ci))
            : (const void*)zerobuf;     // halo pad: copy zeros (keeps issue count uniform)
        async_cp16(&As[s][arow * 32 + akc], asrc);
        const __bf16* bsrc = wBf + ((size_t)(kk >> 5) * 32 + (tileN >> 4)) * 512 + tid * 16;
        async_cp16(&Bs[s][tid * 16],     bsrc);
        async_cp16(&Bs[s][tid * 16 + 8], bsrc + 8);
    };

    const int NSTEP = KTOT / 32;        // 144
    issue(0, 0);
    for (int step = 0; step < NSTEP; ++step) {
        int s = step & 1;
        if (step + 1 < NSTEP) {
            issue(step + 1, s ^ 1);     // prefetch next stage into other buffer
            asm volatile("s_wait_asynccnt 0x3" ::: "memory");  // stage s landed
        } else {
            asm volatile("s_wait_asynccnt 0x0" ::: "memory");
        }
        if (step + 2 < NSTEP)
            __builtin_prefetch((const void*)(wBf + ((size_t)(step + 2) * 32 + (tileN >> 4)) * 512 + tid * 16), 0, 3);
        __syncthreads();

        // A fragment: two contiguous 16B runs per lane -> 2 x ds_load_b128
        const uint4* ap4 = (const uint4*)(&As[s][(wm * 16 + l16) * 32 + half * 16]);
        U32x2 atmp; atmp.a = ap4[0]; atmp.b = ap4[1];
        v16bf afrag = __builtin_bit_cast(v16bf, atmp);

        #pragma unroll
        for (int t = 0; t < 4; ++t) {
            // B fragment: one contiguous 32B run per lane (pre-swizzled)
            v16bf bfrag = *(const v16bf*)(&Bs[s][((wn * 4 + t) * 32 + lane) * 16]);
            acc[t] = __builtin_amdgcn_wmma_f32_16x16x32_bf16(
                false, afrag, false, bfrag, (short)0, acc[t], false, false);
        }
        __syncthreads();
    }

    // ---- write out (+bias) as bf16 [pixel][cout] ----
    #pragma unroll
    for (int t = 0; t < 4; ++t) {
        int cout = tileN + wn * 64 + t * 16 + l16;
        float bv = bias[cout];
        #pragma unroll
        for (int r = 0; r < 8; ++r) {
            int pix = tileM + wm * 16 + half * 8 + r;
            outT[(size_t)pix * COUT + cout] = f2bf(acc[t][r] + bv);
        }
    }
}

// ---------------- small WMMA GEMM : fused 1x1 convs (N=64) ----------------
// BM=128, async double-buffered A, B fragments straight from global (L2-resident)
__global__ __launch_bounds__(256) void k_gemm_small(
    const __bf16* __restrict__ rpnT,    // [NPIX][512]
    const __bf16* __restrict__ w2f,     // swizzled fragments
    const float*  __restrict__ bias2,   // [64]
    float* __restrict__ logits)         // [NPIX][64]
{
    __shared__ __attribute__((aligned(32))) __bf16 As[2][128 * 32];      // 2 x 8 KB
    const int tid  = threadIdx.x;
    const int lane = tid & 31;
    const int wave = tid >> 5;          // wm 0..7
    const int half = lane >> 4;
    const int l16  = lane & 15;
    const int tileM = blockIdx.x * 128;

    v8f acc[4] = {};

    const int arow = tid >> 1;          // 0..127
    const int akc  = (tid & 1) * 16;    // 0/16

    auto issue = [&](int step, int s) {
        const __bf16* src = rpnT + (size_t)(tileM + arow) * CIN + step * 32 + akc;
        async_cp16(&As[s][arow * 32 + akc],     src);
        async_cp16(&As[s][arow * 32 + akc + 8], src + 8);
    };

    issue(0, 0);
    for (int step = 0; step < 16; ++step) {
        int s = step & 1;
        if (step + 1 < 16) {
            issue(step + 1, s ^ 1);
            asm volatile("s_wait_asynccnt 0x2" ::: "memory");
        } else {
            asm volatile("s_wait_asynccnt 0x0" ::: "memory");
        }
        __syncthreads();

        const uint4* ap4 = (const uint4*)(&As[s][(wave * 16 + l16) * 32 + half * 16]);
        U32x2 atmp; atmp.a = ap4[0]; atmp.b = ap4[1];
        v16bf afrag = __builtin_bit_cast(v16bf, atmp);

        #pragma unroll
        for (int t = 0; t < 4; ++t) {
            v16bf bfrag = *(const v16bf*)(w2f + ((size_t)(step * 4 + t) * 32 + lane) * 16);
            acc[t] = __builtin_amdgcn_wmma_f32_16x16x32_bf16(
                false, afrag, false, bfrag, (short)0, acc[t], false, false);
        }
        __syncthreads();
    }
    #pragma unroll
    for (int t = 0; t < 4; ++t) {
        int n = t * 16 + l16;
        float bv = bias2[n];
        #pragma unroll
        for (int r = 0; r < 8; ++r) {
            int pix = tileM + wave * 16 + half * 8 + r;
            logits[(size_t)pix * 64 + n] = acc[t][r] + bv;
        }
    }
}

// ---------------- decode: softmax fg prob + bbox transform + clip ----------------
__global__ void k_decode(const float* __restrict__ logits, const int* __restrict__ im_info,
                         float* __restrict__ boxes, float* __restrict__ scoref,
                         unsigned* __restrict__ skey) {
    int i = blockIdx.x * 256 + threadIdx.x;
    if (i >= NANCH) return;
    int a = i % 9, p = i / 9;
    int x = p & 127, y = p >> 7;
    const float* row = logits + (size_t)p * 64;
    float l0 = row[a], l1 = row[9 + a];
    float fg = 1.0f / (1.0f + __expf(l0 - l1));
    float dx = row[18 + 4 * a + 0];
    float dy = row[18 + 4 * a + 1];
    float dw = row[18 + 4 * a + 2];
    float dh = row[18 + 4 * a + 3];
    const float BC = 4.135166556742356f;   // log(1000/16)
    dw = fminf(fmaxf(dw, -BC), BC);
    dh = fminf(fmaxf(dh, -BC), BC);
    float ax1 = BASE_ANCH[a][0] + 16.f * x;
    float ay1 = BASE_ANCH[a][1] + 16.f * y;
    float ax2 = BASE_ANCH[a][2] + 16.f * x;
    float ay2 = BASE_ANCH[a][3] + 16.f * y;
    float aw = ax2 - ax1 + 1.f, ah = ay2 - ay1 + 1.f;
    float acx = ax1 + 0.5f * aw, acy = ay1 + 0.5f * ah;
    float pcx = dx * aw + acx, pcy = dy * ah + acy;
    float pw = __expf(dw) * aw, ph = __expf(dh) * ah;
    float imw = (float)im_info[1] - 1.f, imh = (float)im_info[0] - 1.f;
    float x1 = fminf(fmaxf(pcx - 0.5f * pw, 0.f), imw);
    float y1 = fminf(fmaxf(pcy - 0.5f * ph, 0.f), imh);
    float x2 = fminf(fmaxf(pcx + 0.5f * pw, 0.f), imw);
    float y2 = fminf(fmaxf(pcy + 0.5f * ph, 0.f), imh);
    boxes[(size_t)i * 4 + 0] = x1;
    boxes[(size_t)i * 4 + 1] = y1;
    boxes[(size_t)i * 4 + 2] = x2;
    boxes[(size_t)i * 4 + 3] = y2;
    scoref[i] = fg;
    skey[i] = __builtin_bit_cast(unsigned, fg);   // positive float -> order-preserving bits
}

// ---------------- top-6000 selection ----------------
__global__ void k_hist(const unsigned* __restrict__ skey, unsigned* __restrict__ hist) {
    int i = blockIdx.x * 256 + threadIdx.x;
    if (i >= NANCH) return;
    atomicAdd(&hist[skey[i] >> 21], 1u);
}

// single WG: find threshold bucket, then deterministic scan-compaction into pairs[8192]
__global__ __launch_bounds__(1024) void k_select(const unsigned* __restrict__ skey,
                                                 const unsigned* __restrict__ hist,
                                                 unsigned* __restrict__ pk,
                                                 unsigned* __restrict__ pi) {
    __shared__ unsigned sh[1024];
    __shared__ unsigned sscan[1024];
    __shared__ unsigned sT, sBase, runA, runE;
    int tid = threadIdx.x;
    sh[tid] = hist[tid];
    __syncthreads();
    sscan[tid] = sh[1023 - tid];                     // reversed for suffix sum
    __syncthreads();
    for (int off = 1; off < 1024; off <<= 1) {
        unsigned v = sscan[tid];
        unsigned add = (tid >= off) ? sscan[tid - off] : 0u;
        __syncthreads();
        sscan[tid] = v + add;
        __syncthreads();
    }
    unsigned cumGE = sscan[1023 - tid];              // count of keys in buckets >= tid
    unsigned cumA  = cumGE - sh[tid];                // strictly above
    if (cumA < PRE_TOPN && cumGE >= PRE_TOPN) { sT = tid; sBase = cumA; }
    if (tid == 0) { runA = 0; runE = 0; }
    __syncthreads();
    unsigned T = sT, base = sBase;

    for (int chunk = 0; chunk < NANCH / 1024; ++chunk) {
        int i = chunk * 1024 + tid;
        unsigned key = skey[i];
        unsigned b = key >> 21;
        unsigned fa = (b > T) ? 1u : 0u;
        unsigned fe = (b == T) ? 1u : 0u;
        unsigned packed = fa | (fe << 16);
        sscan[tid] = packed;
        __syncthreads();
        for (int off = 1; off < 1024; off <<= 1) {
            unsigned v = sscan[tid];
            unsigned add = (tid >= off) ? sscan[tid - off] : 0u;
            __syncthreads();
            sscan[tid] = v + add;
            __syncthreads();
        }
        unsigned incl = sscan[tid];
        unsigned tot  = sscan[1023];
        unsigned exA = (incl & 0xFFFFu) - fa;
        unsigned exE = (incl >> 16) - fe;
        unsigned rA = runA, rE = runE;
        if (fa) { unsigned r = rA + exA;        if (r < PRE_TOPN) { pk[r] = key; pi[r] = (unsigned)i; } }
        if (fe) { unsigned r = base + rE + exE; if (r < PRE_TOPN) { pk[r] = key; pi[r] = (unsigned)i; } }
        __syncthreads();
        if (tid == 0) { runA += (tot & 0xFFFFu); runE += (tot >> 16); }
        __syncthreads();
    }
    for (int j = PRE_TOPN + tid; j < SORTN; j += 1024) { pk[j] = 0u; pi[j] = 0xFFFFFFFFu; }
}

// single WG bitonic sort of 8192 composite keys in 64KB LDS (desc key, asc idx)
__global__ __launch_bounds__(1024) void k_sort(const unsigned* __restrict__ pk,
                                               const unsigned* __restrict__ pi,
                                               const float* __restrict__ boxes,
                                               const float* __restrict__ scoref,
                                               float* __restrict__ sboxes,
                                               float* __restrict__ ssc) {
    __shared__ unsigned long long c[SORTN];   // 64 KB (CDNA5: 320 KB/WGP)
    int tid = threadIdx.x;
    for (int e = 0; e < 8; ++e) {
        int i = e * 1024 + tid;
        c[i] = ((unsigned long long)pk[i] << 32) | (unsigned long long)(0xFFFFFFFFu - pi[i]);
    }
    __syncthreads();
    for (int k = 2; k <= SORTN; k <<= 1) {
        for (int j = k >> 1; j > 0; j >>= 1) {
            for (int e = 0; e < 8; ++e) {
                int i = e * 1024 + tid;
                int ixj = i ^ j;
                if (ixj > i) {
                    bool up = ((i & k) == 0);
                    unsigned long long a = c[i], b = c[ixj];
                    bool sw = up ? (a < b) : (a > b);   // overall descending
                    if (sw) { c[i] = b; c[ixj] = a; }
                }
            }
            __syncthreads();
        }
    }
    for (int e = 0; e < 8; ++e) {
        int r = e * 1024 + tid;
        if (r < PRE_TOPN) {
            unsigned long long v = c[r];
            unsigned idx = 0xFFFFFFFFu - (unsigned)(v & 0xFFFFFFFFu);
            ssc[r] = scoref[idx];
            sboxes[(size_t)r * 4 + 0] = boxes[(size_t)idx * 4 + 0];
            sboxes[(size_t)r * 4 + 1] = boxes[(size_t)idx * 4 + 1];
            sboxes[(size_t)r * 4 + 2] = boxes[(size_t)idx * 4 + 2];
            sboxes[(size_t)r * 4 + 3] = boxes[(size_t)idx * 4 + 3];
        }
    }
}

// ---------------- NMS ----------------
__global__ void k_mask(const float* __restrict__ sboxes, unsigned* __restrict__ mask) {
    int blk = blockIdx.x;
    int i = blk / NWORDS;
    int w = blk % NWORDS;
    int col = w * 32 + threadIdx.x;
    float bx1 = sboxes[(size_t)i * 4 + 0], by1 = sboxes[(size_t)i * 4 + 1];
    float bx2 = sboxes[(size_t)i * 4 + 2], by2 = sboxes[(size_t)i * 4 + 3];
    float barea = (bx2 - bx1 + 1.f) * (by2 - by1 + 1.f);
    bool kill = false;
    if (col < PRE_TOPN && col > i) {
        float cx1 = sboxes[(size_t)col * 4 + 0], cy1 = sboxes[(size_t)col * 4 + 1];
        float cx2 = sboxes[(size_t)col * 4 + 2], cy2 = sboxes[(size_t)col * 4 + 3];
        float iw = fminf(bx2, cx2) - fmaxf(bx1, cx1) + 1.f;
        float ih = fminf(by2, cy2) - fmaxf(by1, cy1) + 1.f;
        iw = fmaxf(iw, 0.f); ih = fmaxf(ih, 0.f);
        float inter = iw * ih;
        float carea = (cx2 - cx1 + 1.f) * (cy2 - cy1 + 1.f);
        float iou = inter / (barea + carea - inter);
        kill = iou > 0.7f;
    }
    unsigned long long b = __ballot(kill);   // wave32 -> low 32 bits
    if (threadIdx.x == 0) mask[(size_t)i * NWORDS + w] = (unsigned)b;
}

__global__ __launch_bounds__(256) void k_nms_reduce(const unsigned* __restrict__ mask,
                                                    unsigned* __restrict__ keep) {
    __shared__ unsigned rem[NWORDS];
    __shared__ int curKeep;
    int tid = threadIdx.x;
    for (int w = tid; w < NWORDS; w += 256) rem[w] = 0u;
    __syncthreads();
    for (int i = 0; i < PRE_TOPN; ++i) {
        if (tid == 0) curKeep = !((rem[i >> 5] >> (i & 31)) & 1u);
        __syncthreads();
        if (curKeep)
            for (int w = tid; w < NWORDS; w += 256) rem[w] |= mask[(size_t)i * NWORDS + w];
        if (tid == 0) keep[i] = (unsigned)curKeep;
        __syncthreads();
    }
}

__global__ __launch_bounds__(1024) void k_final(const unsigned* __restrict__ keep,
                                                const float* __restrict__ sboxes,
                                                const float* __restrict__ ssc,
                                                unsigned* __restrict__ ranks,
                                                float* __restrict__ out) {
    __shared__ unsigned sscan[1024];
    __shared__ unsigned runK, runS;
    int tid = threadIdx.x;
    if (tid == 0) { runK = 0; runS = 0; }
    __syncthreads();
    for (int chunk = 0; chunk < 6; ++chunk) {
        int i = chunk * 1024 + tid;
        unsigned kp = (i < PRE_TOPN) ? keep[i] : 0u;
        unsigned sp = (i < PRE_TOPN) ? (1u - kp) : 0u;
        unsigned packed = kp | (sp << 16);
        sscan[tid] = packed;
        __syncthreads();
        for (int off = 1; off < 1024; off <<= 1) {
            unsigned v = sscan[tid];
            unsigned add = (tid >= off) ? sscan[tid - off] : 0u;
            __syncthreads();
            sscan[tid] = v + add;
            __syncthreads();
        }
        unsigned incl = sscan[tid], tot = sscan[1023];
        unsigned exK = (incl & 0xFFFFu) - kp;
        unsigned exS = (incl >> 16) - sp;
        unsigned rK = runK, rS = runS;
        if (i < PRE_TOPN) ranks[i] = (rK + exK) | ((rS + exS) << 16);
        __syncthreads();
        if (tid == 0) { runK += tot & 0xFFFFu; runS += tot >> 16; }
        __syncthreads();
    }
    unsigned K = runK;
    __syncthreads();
    for (int i = tid; i < PRE_TOPN; i += 1024) {
        unsigned rk = ranks[i];
        unsigned slot = keep[i] ? (rk & 0xFFFFu) : (K + (rk >> 16));
        if (slot < POST_TOPN) {
            out[slot * 5 + 0] = 0.f;
            out[slot * 5 + 1] = sboxes[(size_t)i * 4 + 0];
            out[slot * 5 + 2] = sboxes[(size_t)i * 4 + 1];
            out[slot * 5 + 3] = sboxes[(size_t)i * 4 + 2];
            out[slot * 5 + 4] = sboxes[(size_t)i * 4 + 3];
            out[1500 + slot] = ssc[i];
        }
    }
}

// ---------------- launcher ----------------
extern "C" void kernel_launch(void* const* d_in, const int* in_sizes, int n_in,
                              void* d_out, int out_size, void* d_ws, size_t ws_size,
                              hipStream_t stream) {
    (void)in_sizes; (void)n_in; (void)out_size; (void)ws_size;
    const float* feature = (const float*)d_in[0];
    const int*   im_info = (const int*)d_in[1];
    const float* w_rpn   = (const float*)d_in[2];
    const float* b_rpn   = (const float*)d_in[3];
    const float* w_score = (const float*)d_in[4];
    const float* b_score = (const float*)d_in[5];
    const float* w_bbox  = (const float*)d_in[6];
    const float* b_bbox  = (const float*)d_in[7];
    float* out = (float*)d_out;

    char* p = (char*)d_ws;
    auto alloc = [&](size_t bytes) -> char* {
        char* r = p; p += (bytes + 255) & ~(size_t)255; return r;
    };
    __bf16*  featT  = (__bf16*)  alloc((size_t)NPIX * CIN * 2);
    __bf16*  wBf    = (__bf16*)  alloc((size_t)KTOT * COUT * 2);
    __bf16*  rpnT   = (__bf16*)  alloc((size_t)NPIX * COUT * 2);
    __bf16*  w2f    = (__bf16*)  alloc((size_t)CIN * 64 * 2);
    float*   bias2  = (float*)   alloc(64 * 4);
    float*   logits = (float*)   alloc((size_t)NPIX * 64 * 4);
    float*   boxes  = (float*)   alloc((size_t)NANCH * 4 * 4);
    float*   scoref = (float*)   alloc((size_t)NANCH * 4);
    unsigned* skey  = (unsigned*)alloc((size_t)NANCH * 4);
    unsigned* hist  = (unsigned*)alloc(1024 * 4);
    __bf16*  zerob  = (__bf16*)  alloc(256);
    unsigned* pk    = (unsigned*)alloc(SORTN * 4);
    unsigned* pi    = (unsigned*)alloc(SORTN * 4);
    float*   sboxes = (float*)   alloc((size_t)PRE_TOPN * 4 * 4);
    float*   ssc    = (float*)   alloc((size_t)PRE_TOPN * 4);
    unsigned* mask  = (unsigned*)alloc((size_t)PRE_TOPN * NWORDS * 4);
    unsigned* keep  = (unsigned*)alloc((size_t)PRE_TOPN * 4);
    unsigned* ranks = (unsigned*)alloc((size_t)PRE_TOPN * 4);

    // init (zero pad source must precede GEMM)
    k_init<<<4, 256, 0, stream>>>(hist, zerob);

    // prep
    k_prep_feat<<<(NPIX * CIN + 255) / 256, 256, 0, stream>>>(feature, featT);
    k_prep_wconv<<<(KTOT * COUT + 255) / 256, 256, 0, stream>>>(w_rpn, wBf);
    k_prep_w2<<<(CIN * 64 + 255) / 256, 256, 0, stream>>>(w_score, b_score, w_bbox, b_bbox, w2f, bias2);

    // 3x3 conv as WMMA implicit GEMM (async double-buffered)
    dim3 g1(NPIX / 64, COUT / 128);
    k_conv_gemm<<<g1, 256, 0, stream>>>(featT, wBf, b_rpn, zerob, rpnT);

    // fused score+bbox 1x1 convs
    k_gemm_small<<<NPIX / 128, 256, 0, stream>>>(rpnT, w2f, bias2, logits);

    // decode + keys
    k_decode<<<(NANCH + 255) / 256, 256, 0, stream>>>(logits, im_info, boxes, scoref, skey);

    // top-6000 via histogram threshold + deterministic compaction + LDS bitonic sort
    k_hist<<<(NANCH + 255) / 256, 256, 0, stream>>>(skey, hist);
    k_select<<<1, 1024, 0, stream>>>(skey, hist, pk, pi);
    k_sort<<<1, 1024, 0, stream>>>(pk, pi, boxes, scoref, sboxes, ssc);

    // NMS
    k_mask<<<PRE_TOPN * NWORDS, 32, 0, stream>>>(sboxes, mask);
    k_nms_reduce<<<1, 256, 0, stream>>>(mask, keep);
    k_final<<<1, 1024, 0, stream>>>(keep, sboxes, ssc, ranks, out);

    // feature passthrough (3rd output)
    hipMemcpyAsync(out + 1800, feature, (size_t)CIN * NPIX * sizeof(float),
                   hipMemcpyDeviceToDevice, stream);
}
